// CHMLearner_2388001816579
// MI455X (gfx1250) — matrix operand order repaired and checked by reading, not compiled, not run on hardware
//
#include <hip/hip_runtime.h>

typedef __attribute__((ext_vector_type(16))) _Float16 v16h;
typedef __attribute__((ext_vector_type(8)))  _Float16 v8h;
typedef __attribute__((ext_vector_type(8)))  float    v8f;

// ---------------------------------------------------------------- helpers
__device__ __forceinline__ void ac_coord(int o, int n_out, int n_in,
                                         int& i0, int& i1, float& w) {
  float c = (n_out > 1) ? (float)o * (float)(n_in - 1) / (float)(n_out - 1) : 0.f;
  int f = (int)floorf(c);
  if (f < 0) f = 0;
  if (f > n_in - 1) f = n_in - 1;
  i0 = f;
  i1 = (f + 1 < n_in) ? f + 1 : n_in - 1;
  w = c - (float)f;
}

// ----------------------------------- bilinear resize + f16 pack with zero halo
// in: f32 NCHW [B][C][Hin][Hin] ; out: f16 [B][(s+2)*(s+2)][C], 1-px zero halo.
// Halo-in-memory removes all per-lane padding predication from the conv GEMM.
__global__ void resize_pack_pad(const float* __restrict__ in,
                                _Float16* __restrict__ out,
                                int B, int C, int Hin, int s) {
  long idx = blockIdx.x * (long)blockDim.x + threadIdx.x;
  int sp = s + 2;
  long total = (long)B * sp * sp * C;
  if (idx >= total) return;
  int c = (int)(idx % C);
  long p = idx / C;
  int xp = (int)(p % sp); p /= sp;
  int yp = (int)(p % sp);
  int b = (int)(p / sp);
  float v = 0.f;
  if (xp >= 1 && xp <= s && yp >= 1 && yp <= s) {
    int y0, y1, x0, x1; float wy, wx;
    ac_coord(yp - 1, s, Hin, y0, y1, wy);
    ac_coord(xp - 1, s, Hin, x0, x1, wx);
    const float* base = in + ((long)b * C + c) * Hin * Hin;
    v = (1.f - wy) * ((1.f - wx) * base[y0 * Hin + x0] + wx * base[y0 * Hin + x1]) +
        wy * ((1.f - wx) * base[y1 * Hin + x0] + wx * base[y1 * Hin + x1]);
  }
  out[idx] = (_Float16)v;
}

// w [OC][IC][3][3] f32 -> out [kk][oc][ic] f16  (K-contiguous rows = B fragments)
__global__ void pack_weights_half(const float* __restrict__ w,
                                  _Float16* __restrict__ out, int OC, int IC) {
  long idx = blockIdx.x * (long)blockDim.x + threadIdx.x;
  long total = 9L * OC * IC;
  if (idx >= total) return;
  int ic = (int)(idx % IC);
  long r = idx / IC;
  int oc = (int)(r % OC);
  int kk = (int)(r / OC);
  out[idx] = (_Float16)w[((long)oc * IC + ic) * 9 + kk];
}

// ---------------------------------------------- conv 3x3 as implicit GEMM (WMMA)
// fpad [B][(s+2)^2][1024] f16 (zero halo) ; wpk [9][256][1024] f16 ;
// out [B][s*s][256] f16.  Wave tile: 16x32 (two N tiles share the A fragment).
// Fragments loaded straight from global (operands are L2-resident; 192MB L2):
//   A lane: row m=lane&15, two 8-half runs at k = ic0+16*hi / ic0+16*hi+8
//   B lane: col oc, 16 contiguous halves at k = ic0+16*hi
__global__ void __launch_bounds__(128)
conv3x3_wmma(const _Float16* __restrict__ fpad,
             const _Float16* __restrict__ wpk,
             _Float16* __restrict__ out,
             int B, int s) {
  const int IC = 1024, OC = 256;
  const int M = s * s, SP = s + 2;
  const int mTiles = M / 16, nBlks = OC / 128;
  int bid = blockIdx.x;
  int nb = bid % nBlks; bid /= nBlks;
  int mt = bid % mTiles;
  int b  = bid / mTiles;
  int mBase = mt * 16, nBase = nb * 128;

  int t = threadIdx.x, lane = t & 31, wave = t >> 5;
  int hi = lane >> 4;
  int mrow = mBase + (lane & 15);
  int y = mrow / s, x = mrow % s;
  int oc0 = nBase + wave * 32 + (lane & 15);
  int oc1 = oc0 + 16;
  const _Float16* fb = fpad + (long)b * SP * SP * IC;
  v8f acc0 = {}, acc1 = {};

  for (int kk = 0; kk < 9; ++kk) {
    int dy = kk / 3, dx = kk % 3;
    const _Float16* arow = fb + ((long)(y + dy) * SP + (x + dx)) * IC;
    const _Float16* brow0 = wpk + ((long)kk * OC + oc0) * IC;
    const _Float16* brow1 = wpk + ((long)kk * OC + oc1) * IC;
    __builtin_prefetch((const void*)(brow0 + (long)OC * IC), 0, 3); // next tap
    #pragma unroll 8
    for (int ic0 = 0; ic0 < IC; ic0 += 32) {
      v8h alo = *(const v8h*)(arow + ic0 + 16 * hi);
      v8h ahi = *(const v8h*)(arow + ic0 + 16 * hi + 8);
      v16h af = __builtin_shufflevector(alo, ahi,
                                        0, 1, 2, 3, 4, 5, 6, 7,
                                        8, 9, 10, 11, 12, 13, 14, 15);
      v16h bf0 = *(const v16h*)(brow0 + ic0 + 16 * hi);
      v16h bf1 = *(const v16h*)(brow1 + ic0 + 16 * hi);
      acc0 = __builtin_amdgcn_wmma_f32_16x16x32_f16(false, af, false, bf0,
                                                    (short)0, acc0, false, false);
      acc1 = __builtin_amdgcn_wmma_f32_16x16x32_f16(false, af, false, bf1,
                                                    (short)0, acc1, false, false);
    }
  }
  #pragma unroll
  for (int r = 0; r < 8; ++r) { // C: VGPR r -> M = r + 8*hi
    int p = mBase + r + (hi << 3);
    long base = ((long)b * M + p) * OC;
    out[base + oc0] = (_Float16)acc0[r];
    out[base + oc1] = (_Float16)acc1[r];
  }
}

// ------------------------------------------------------------- norms (1/||.||)
__global__ void inv_norm_rows(const _Float16* __restrict__ a, float* __restrict__ out,
                              int B, int P, int C) { // a [B][P][C]
  int idx = blockIdx.x * blockDim.x + threadIdx.x;
  if (idx >= B * P) return;
  const _Float16* row = a + (long)idx * C;
  float s = 0.f;
  for (int c = 0; c < C; ++c) { float v = (float)row[c]; s += v * v; }
  float n = sqrtf(s);
  out[idx] = (n > 0.f) ? 1.f / n : 0.f;
}

// --------------------------------------------- correlation GEMM (WMMA, K=256)
// A [B][Ms][256] f16 (src conv) ; Bt [B][Nt][256] f16 (trg conv, pos-major) ;
// out [B][Ms][Nt] f32 cosine sims. Wave tile: 32x16 (two M tiles share the B
// fragment). Fully unrolled K -> 16 back-to-back WMMAs.
__global__ void __launch_bounds__(128)
corr_wmma(const _Float16* __restrict__ A, const _Float16* __restrict__ Bt,
          const float* __restrict__ invs, const float* __restrict__ invt,
          float* __restrict__ out, int B, int Ms, int Nt) {
  const int K = 256;
  const int mTiles = Ms / 32, nBlks = Nt / 64;
  int bid = blockIdx.x;
  int nb = bid % nBlks; bid /= nBlks;
  int mt = bid % mTiles;
  int b  = bid / mTiles;
  int mBase = mt * 32, nBase = nb * 64;

  int t = threadIdx.x, lane = t & 31, wave = t >> 5;
  int hi = lane >> 4;
  const _Float16* arow0 = A + ((long)b * Ms + mBase + (lane & 15)) * K;
  const _Float16* arow1 = arow0 + 16 * K;
  int n = nBase + wave * 16 + (lane & 15);
  const _Float16* brow = Bt + ((long)b * Nt + n) * K;
  v8f acc0 = {}, acc1 = {};

  #pragma unroll
  for (int ic0 = 0; ic0 < K; ic0 += 32) {
    v16h bf = *(const v16h*)(brow + ic0 + 16 * hi);
    v8h a0lo = *(const v8h*)(arow0 + ic0 + 16 * hi);
    v8h a0hi = *(const v8h*)(arow0 + ic0 + 16 * hi + 8);
    v16h af0 = __builtin_shufflevector(a0lo, a0hi,
                                       0, 1, 2, 3, 4, 5, 6, 7,
                                       8, 9, 10, 11, 12, 13, 14, 15);
    v8h a1lo = *(const v8h*)(arow1 + ic0 + 16 * hi);
    v8h a1hi = *(const v8h*)(arow1 + ic0 + 16 * hi + 8);
    v16h af1 = __builtin_shufflevector(a1lo, a1hi,
                                       0, 1, 2, 3, 4, 5, 6, 7,
                                       8, 9, 10, 11, 12, 13, 14, 15);
    acc0 = __builtin_amdgcn_wmma_f32_16x16x32_f16(false, af0, false, bf,
                                                  (short)0, acc0, false, false);
    acc1 = __builtin_amdgcn_wmma_f32_16x16x32_f16(false, af1, false, bf,
                                                  (short)0, acc1, false, false);
  }
  float it = invt[(long)b * Nt + n];
  #pragma unroll
  for (int r = 0; r < 8; ++r) {
    int p0 = mBase + r + (hi << 3);
    int p1 = p0 + 16;
    out[((long)b * Ms + p0) * Nt + n] = acc0[r] * invs[(long)b * Ms + p0] * it;
    out[((long)b * Ms + p1) * Nt + n] = acc1[r] * invs[(long)b * Ms + p1] * it;
  }
}

// ------------------------------------------- interpolate4d passes (align_corners)
// pass1: in [B][H*W][C] resize (H,W)->(OH,OW) : out [B][OH*OW][C]
__global__ void resize_rows(const float* __restrict__ in, float* __restrict__ out,
                            int B, int H, int W, int C, int OH, int OW) {
  long idx = blockIdx.x * (long)blockDim.x + threadIdx.x;
  long total = (long)B * OH * OW * C;
  if (idx >= total) return;
  int c = (int)(idx % C);
  long p = idx / C;
  int ox = (int)(p % OW); p /= OW;
  int oy = (int)(p % OH);
  int b = (int)(p / OH);
  int y0, y1, x0, x1; float wy, wx;
  ac_coord(oy, OH, H, y0, y1, wy);
  ac_coord(ox, OW, W, x0, x1, wx);
  const float* base = in + (long)b * H * W * C;
  float v = (1.f - wy) * ((1.f - wx) * base[((long)y0 * W + x0) * C + c] +
                          wx * base[((long)y0 * W + x1) * C + c]) +
            wy * ((1.f - wx) * base[((long)y1 * W + x0) * C + c] +
                  wx * base[((long)y1 * W + x1) * C + c]);
  out[idx] = v;
}
// pass2: in [B][C][H*W] resize inner (H,W)->(OH,OW): out[b*outBStride + c*OH*OW + p]
__global__ void resize_cols(const float* __restrict__ in, float* __restrict__ out,
                            int B, int C, int H, int W, int OH, int OW,
                            int relu, long outBStride) {
  long idx = blockIdx.x * (long)blockDim.x + threadIdx.x;
  long total = (long)B * C * OH * OW;
  if (idx >= total) return;
  int pw = (int)(idx % OW);
  long r = idx / OW;
  int ph = (int)(r % OH); r /= OH;
  int c = (int)(r % C);
  int b = (int)(r / C);
  int y0, y1, x0, x1; float wy, wx;
  ac_coord(ph, OH, H, y0, y1, wy);
  ac_coord(pw, OW, W, x0, x1, wx);
  const float* base = in + ((long)b * C + c) * H * W;
  float v = (1.f - wy) * ((1.f - wx) * base[y0 * W + x0] + wx * base[y0 * W + x1]) +
            wy * ((1.f - wx) * base[y1 * W + x0] + wx * base[y1 * W + x1]);
  if (relu) v = fmaxf(v, 0.f);
  out[(long)b * outBStride + ((long)c * OH * OW + (long)ph * OW + pw)] = v;
}

// -------------------------------------------------- chm6d: 3x3x5^4 6D stencil
// corr6 [B][9][16][16][16][16], k6 [3][3][5][5][5][5], out same shape
__global__ void __launch_bounds__(256)
chm6d_kernel(const float* __restrict__ corr6, const float* __restrict__ k6,
             float* __restrict__ out) {
  __shared__ float lk[5625];
  __shared__ float tile[400]; // 20x20 zero-padded (u,v) plane
  int t = threadIdx.x;
  for (int i = t; i < 5625; i += 256) lk[i] = k6[i];
  int bid = blockIdx.x;
  int x = bid & 15; bid >>= 4;
  int y = bid & 15; bid >>= 4;
  int sj = bid % 3; bid /= 3;
  int si = bid % 3;
  int b = bid / 3;
  int u = t >> 4, v = t & 15;
  float acc = 0.f;
  for (int ki = 0; ki < 3; ++ki) {
    int si2 = si + ki - 1;
    if (si2 < 0 || si2 >= 3) continue;
    for (int kj = 0; kj < 3; ++kj) {
      int sj2 = sj + kj - 1;
      if (sj2 < 0 || sj2 >= 3) continue;
      const float* plane00 = corr6 + (((long)b * 9 + si2 * 3 + sj2) << 16);
      const float* kker = lk + (ki * 3 + kj) * 625;
      for (int a = 0; a < 5; ++a) {
        int y2 = y + a - 2;
        if (y2 < 0 || y2 >= 16) continue;
        for (int bb = 0; bb < 5; ++bb) {
          int x2 = x + bb - 2;
          if (x2 < 0 || x2 >= 16) continue;
          const float* pl = plane00 + ((y2 * 16 + x2) << 8);
          __syncthreads();
          for (int i = t; i < 400; i += 256) {
            int uu = i / 20 - 2, vv = i % 20 - 2;
            tile[i] = (uu >= 0 && uu < 16 && vv >= 0 && vv < 16) ? pl[uu * 16 + vv] : 0.f;
          }
          __syncthreads();
          const float* kab = kker + (a * 5 + bb) * 25;
          #pragma unroll
          for (int c = 0; c < 5; ++c)
            #pragma unroll
            for (int d = 0; d < 5; ++d)
              acc += tile[(u + c) * 20 + (v + d)] * kab[c * 5 + d];
        }
      }
    }
  }
  out[((((long)b * 9 + si * 3 + sj) << 8) + (y * 16 + x)) * 256 + u * 16 + v] = acc;
}

// sigmoid then max over 9 scale pairs: in [B][9][65536] -> out [B][65536]
__global__ void sigmoid_max9(const float* __restrict__ in, float* __restrict__ out, int B) {
  long idx = blockIdx.x * (long)blockDim.x + threadIdx.x;
  long total = (long)B * 65536;
  if (idx >= total) return;
  int p = (int)(idx & 65535);
  int b = (int)(idx >> 16);
  const float* base = in + ((long)b * 9) * 65536 + p;
  float m = -1e30f;
  for (int k = 0; k < 9; ++k) {
    float v = base[(long)k * 65536];
    v = 1.f / (1.f + expf(-v));
    m = fmaxf(m, v);
  }
  out[idx] = m;
}

// ---------------------------------------------- fast4d 5^4 + bias + softplus
// in [B][32][32][32][32], out (cm) [B][1024][1024]
__global__ void __launch_bounds__(256)
fast4d_kernel(const float* __restrict__ in, const float* __restrict__ k4,
              const float* __restrict__ bias, float* __restrict__ out) {
  __shared__ float lk[625];
  __shared__ float tile[1296]; // 36x36 zero-padded (u,v) plane
  int t = threadIdx.x;
  for (int i = t; i < 625; i += 256) lk[i] = k4[i];
  int bid = blockIdx.x;
  int x = bid & 31; bid >>= 5;
  int y = bid & 31;
  int b = bid >> 5;
  float acc[4] = {0.f, 0.f, 0.f, 0.f};
  for (int a = 0; a < 5; ++a) {
    int y2 = y + a - 2;
    if (y2 < 0 || y2 >= 32) continue;
    for (int bb = 0; bb < 5; ++bb) {
      int x2 = x + bb - 2;
      if (x2 < 0 || x2 >= 32) continue;
      const float* pl = in + (((long)b * 1024 + y2 * 32 + x2) << 10);
      __syncthreads();
      for (int i = t; i < 1296; i += 256) {
        int uu = i / 36 - 2, vv = i % 36 - 2;
        tile[i] = (uu >= 0 && uu < 32 && vv >= 0 && vv < 32) ? pl[uu * 32 + vv] : 0.f;
      }
      __syncthreads();
      const float* kab = lk + (a * 5 + bb) * 25;
      #pragma unroll
      for (int q = 0; q < 4; ++q) {
        int uv = t + q * 256;
        int u = uv >> 5, v = uv & 31;
        float s = 0.f;
        #pragma unroll
        for (int c = 0; c < 5; ++c)
          #pragma unroll
          for (int d = 0; d < 5; ++d)
            s += tile[(u + c) * 36 + (v + d)] * kab[c * 5 + d];
        acc[q] += s;
      }
    }
  }
  float bv = bias[0];
  long rowBase = ((long)b * 1024 + (y * 32 + x)) * 1024;
  #pragma unroll
  for (int q = 0; q < 4; ++q) {
    float v = acc[q] + bv;
    v = (v > 20.f) ? v : log1pf(expf(v)); // softplus
    out[rowBase + t + q * 256] = v;
  }
}

// ------------------------------------------------------- mutual NN filtering
__global__ void row_max(const float* __restrict__ cm, float* __restrict__ rmax, int B) {
  int idx = blockIdx.x * blockDim.x + threadIdx.x;
  if (idx >= B * 1024) return;
  const float* row = cm + (long)idx * 1024;
  float m = row[0];
  for (int j = 1; j < 1024; ++j) m = fmaxf(m, row[j]);
  rmax[idx] = (m == 0.f) ? 1e-30f : m;
}
__global__ void col_max(const float* __restrict__ cm, float* __restrict__ cmax, int B) {
  int idx = blockIdx.x * blockDim.x + threadIdx.x;
  if (idx >= B * 1024) return;
  int c = idx & 1023, b = idx >> 10;
  const float* base = cm + ((long)b << 20) + c;
  float m = base[0];
  for (int r = 1; r < 1024; ++r) m = fmaxf(m, base[(long)r * 1024]);
  cmax[idx] = (m == 0.f) ? 1e-30f : m;
}
__global__ void mutual_kernel(const float* __restrict__ cm, const float* __restrict__ rmax,
                              const float* __restrict__ cmax, float* __restrict__ out, int B) {
  long idx = blockIdx.x * (long)blockDim.x + threadIdx.x;
  long total = (long)B << 20;
  if (idx >= total) return;
  int c = (int)(idx & 1023);
  long r = idx >> 10;
  int rr = (int)(r & 1023);
  int b = (int)(r >> 10);
  float v = cm[idx];
  out[idx] = v * (v / rmax[b * 1024 + rr]) * (v / cmax[b * 1024 + c]);
}

// ================================================================ launcher
extern "C" void kernel_launch(void* const* d_in, const int* in_sizes, int n_in,
                              void* d_out, int out_size, void* d_ws, size_t ws_size,
                              hipStream_t stream) {
  (void)in_sizes; (void)n_in; (void)out_size; (void)ws_size;
  const int B = 4, FD = 1024, SIDE = 16, OC = 256;
  const int S[3] = {8, 16, 32};

  const float* src = (const float*)d_in[0];
  const float* trg = (const float*)d_in[1];
  const float* w[3] = {(const float*)d_in[2], (const float*)d_in[3], (const float*)d_in[4]};
  const float* k6 = (const float*)d_in[5];
  const float* k4 = (const float*)d_in[6];
  const float* b4 = (const float*)d_in[7];

  size_t off = 0;
  auto alloc = [&](size_t bytes) -> void* {
    void* p = (char*)d_ws + off;
    off += (bytes + 255) & ~(size_t)255;
    return p;
  };
  _Float16 *fsrc[3], *ftrg[3], *wpk[3], *csrc[3], *ctrg[3];
  float *invs[3], *invt[3], *corrP[3][3];
  for (int i = 0; i < 3; ++i) {
    size_t n = (size_t)B * (S[i] + 2) * (S[i] + 2) * FD; // padded (halo) features
    fsrc[i] = (_Float16*)alloc(n * 2);
    ftrg[i] = (_Float16*)alloc(n * 2);
  }
  for (int i = 0; i < 3; ++i) wpk[i] = (_Float16*)alloc((size_t)9 * OC * FD * 2);
  for (int i = 0; i < 3; ++i) {
    size_t n = (size_t)B * S[i] * S[i] * OC;
    csrc[i] = (_Float16*)alloc(n * 2);
    ctrg[i] = (_Float16*)alloc(n * 2);
  }
  for (int i = 0; i < 3; ++i) {
    invs[i] = (float*)alloc((size_t)B * S[i] * S[i] * 4);
    invt[i] = (float*)alloc((size_t)B * S[i] * S[i] * 4);
  }
  for (int i = 0; i < 3; ++i)
    for (int j = 0; j < 3; ++j)
      corrP[i][j] = (float*)alloc((size_t)B * S[i] * S[i] * S[j] * S[j] * 4);
  float* tmp   = (float*)alloc((size_t)B * 1024 * 256 * 4); // reused for both interps
  float* corr6 = (float*)alloc((size_t)B * 9 * 65536 * 4);
  float* chmo  = (float*)alloc((size_t)B * 9 * 65536 * 4);
  float* max16 = (float*)alloc((size_t)B * 65536 * 4);
  float* big32 = (float*)alloc((size_t)B * 1024 * 1024 * 4);
  float* cm    = (float*)alloc((size_t)B * 1024 * 1024 * 4);
  float* rmx   = (float*)alloc((size_t)B * 1024 * 4);
  float* cmx   = (float*)alloc((size_t)B * 1024 * 4);

  // 1) resize+pack features (with zero halo), pack weights
  for (int i = 0; i < 3; ++i) {
    long tot = (long)B * (S[i] + 2) * (S[i] + 2) * FD;
    int blks = (int)((tot + 255) / 256);
    resize_pack_pad<<<blks, 256, 0, stream>>>(src, fsrc[i], B, FD, SIDE, S[i]);
    resize_pack_pad<<<blks, 256, 0, stream>>>(trg, ftrg[i], B, FD, SIDE, S[i]);
    long wt = 9L * OC * FD;
    pack_weights_half<<<(int)((wt + 255) / 256), 256, 0, stream>>>(w[i], wpk[i], OC, FD);
  }
  // 2) conv3x3 via WMMA implicit GEMM + norms
  for (int i = 0; i < 3; ++i) {
    int M = S[i] * S[i];
    int grid = B * (M / 16) * (OC / 128);
    conv3x3_wmma<<<grid, 128, 0, stream>>>(fsrc[i], wpk[i], csrc[i], B, S[i]);
    conv3x3_wmma<<<grid, 128, 0, stream>>>(ftrg[i], wpk[i], ctrg[i], B, S[i]);
    int np = B * M;
    inv_norm_rows<<<(np + 255) / 256, 256, 0, stream>>>(csrc[i], invs[i], B, M, OC);
    inv_norm_rows<<<(np + 255) / 256, 256, 0, stream>>>(ctrg[i], invt[i], B, M, OC);
  }
  // 3) 9 correlation GEMMs (WMMA) + interpolate4d to 16^4 (+ReLU) into corr6
  for (int i = 0; i < 3; ++i)
    for (int j = 0; j < 3; ++j) {
      int Ms = S[i] * S[i], Nt = S[j] * S[j];
      int grid = B * (Ms / 32) * (Nt / 64);
      corr_wmma<<<grid, 128, 0, stream>>>(csrc[i], ctrg[j], invs[i], invt[j],
                                          corrP[i][j], B, Ms, Nt);
      long t1 = (long)B * 16 * 16 * Nt;
      resize_rows<<<(int)((t1 + 255) / 256), 256, 0, stream>>>(
          corrP[i][j], tmp, B, S[i], S[i], Nt, 16, 16);
      long t2 = (long)B * 256 * 256;
      resize_cols<<<(int)((t2 + 255) / 256), 256, 0, stream>>>(
          tmp, corr6 + (long)(i * 3 + j) * 65536, B, 256, S[j], S[j], 16, 16,
          /*relu=*/1, /*outBStride=*/(long)9 * 65536);
    }
  // 4) 6D CHM conv, sigmoid + max over 9 pairs
  chm6d_kernel<<<B * 9 * 256, 256, 0, stream>>>(corr6, k6, chmo);
  {
    long t = (long)B * 65536;
    sigmoid_max9<<<(int)((t + 255) / 256), 256, 0, stream>>>(chmo, max16, B);
  }
  // 5) interpolate4d 16^4 -> 32^4
  {
    long t1 = (long)B * 32 * 32 * 256;
    resize_rows<<<(int)((t1 + 255) / 256), 256, 0, stream>>>(max16, tmp, B, 16, 16, 256, 32, 32);
    long t2 = (long)B * 1024 * 1024;
    resize_cols<<<(int)((t2 + 255) / 256), 256, 0, stream>>>(
        tmp, big32, B, 1024, 16, 16, 32, 32, /*relu=*/0, /*outBStride=*/(long)1024 * 1024);
  }
  // 6) 5^4 conv + bias + softplus -> cm
  fast4d_kernel<<<B * 1024, 256, 0, stream>>>(big32, k4, b4, cm);
  // 7) mutual NN filter -> d_out
  row_max<<<(B * 1024 + 255) / 256, 256, 0, stream>>>(cm, rmx, B);
  col_max<<<(B * 1024 + 255) / 256, 256, 0, stream>>>(cm, cmx, B);
  {
    long t = (long)B << 20;
    mutual_kernel<<<(int)((t + 255) / 256), 256, 0, stream>>>(cm, rmx, cmx, (float*)d_out, B);
  }
}
